// AWQWeight_1571958030525
// MI455X (gfx1250) — matrix operand order repaired and checked by reading, not compiled, not run on hardware
//
#include <hip/hip_runtime.h>

typedef __attribute__((ext_vector_type(16))) _Float16 v16h;
typedef __attribute__((ext_vector_type(8)))  _Float16 v8h;
typedef __attribute__((ext_vector_type(2)))  _Float16 v2h;
typedef __attribute__((ext_vector_type(8)))  float    v8f;
typedef __attribute__((ext_vector_type(4)))  float    v4f;

#define BM 256
#define BN 128
#define BK 32
#define LDT 40          // BK + 8 halves pad: row stride 80B (16B aligned, bank-spread)
#define GSZ 128
#define NTHREADS 256

__global__ __launch_bounds__(NTHREADS) void awq_gemm_f16_wmma(
    const float* __restrict__ x,      // [M,K] fp32
    const int*   __restrict__ qw,     // [K,N/8] packed int4 (AWQ order)
    const float* __restrict__ sc,     // [K/G,N] fp32
    const int*   __restrict__ qz,     // [K/G,N/8] packed int4 (AWQ order)
    const float* __restrict__ bias,   // [N]
    float*       __restrict__ out,    // [M,N]
    int M, int K, int N)
{
  __shared__ _Float16 sA[2][BM * LDT];   // [m][k] halves, K-contiguous
  __shared__ _Float16 sB[2][BN * LDT];   // [n][k] halves, K-contiguous

  const int tid  = threadIdx.x;
  const int lane = tid & 31;
  const int wave = tid >> 5;
  const int wm   = wave >> 1;            // 0..3 -> 64 output rows each
  const int wn   = wave & 1;             // 0..1 -> 64 output cols each
  const int bm   = blockIdx.y * BM;
  const int bn   = blockIdx.x * BN;
  const int Np   = N >> 3;               // packed columns

  // global->LDS staging assignments
  const int am = tid;                    // A: one row per thread (256 rows)
  const int bc = tid & 15;               // B: packed col within tile (16/tile)
  const int bk = (tid >> 4) * 2;         // B: k row base (2 rows/thread)

  const int S = K / BK;

  v4f areg[8];   // 32 floats of x (one row, full BK)
  int qreg[2];   // 2 packed weight words (2 k-rows, same column)
  int zreg;      // packed zeros for this column/group
  v4f sreg[2];   // 8 scales for this column/group

  auto loadG = [&](int s) {
    const int k0 = s * BK;
    const float* ap = x + (size_t)(bm + am) * K + k0;
#pragma unroll
    for (int i = 0; i < 8; ++i) areg[i] = *(const v4f*)(ap + 4 * i);
    const int g = k0 / GSZ;
    const int* qp = qw + (size_t)(k0 + bk) * Np + (bn >> 3) + bc;
    qreg[0] = qp[0];
    qreg[1] = qp[Np];
    zreg = qz[(size_t)g * Np + (bn >> 3) + bc];
    const float* sp = sc + (size_t)g * N + bn + bc * 8;
    sreg[0] = *(const v4f*)sp;
    sreg[1] = *(const v4f*)(sp + 4);
  };

  auto storeL = [&](int buf) {
    // A: fp32 -> f16, four aligned b128 LDS stores (one full row of BK)
    _Float16* pa = &sA[buf][am * LDT];
#pragma unroll
    for (int h = 0; h < 4; ++h) {
      v8h v;
#pragma unroll
      for (int e = 0; e < 4; ++e) {
        v[e]     = (_Float16)areg[2 * h][e];
        v[e + 4] = (_Float16)areg[2 * h + 1][e];
      }
      *(v8h*)(pa + 8 * h) = v;
    }
    // B: AWQ unpack + dequant, transpose-in-regs to [n][k] layout
#pragma unroll
    for (int j = 0; j < 8; ++j) {
      const int sh  = 4 * ((j >> 1) + ((j & 1) << 2));  // AWQ_REVERSE_ORDER
      const float z = (float)((zreg >> sh) & 0xF);
      const float s = (j < 4) ? sreg[0][j] : sreg[1][j - 4];
      const float q0 = (float)((qreg[0] >> sh) & 0xF);
      const float q1 = (float)((qreg[1] >> sh) & 0xF);
      v2h v;
      v[0] = (_Float16)((q0 - z) * s);
      v[1] = (_Float16)((q1 - z) * s);
      *(v2h*)(&sB[buf][(bc * 8 + j) * LDT + bk]) = v;   // 4B-aligned b32 store
    }
  };

  v8f acc[4][4];
  {
    v8f zero = {};
#pragma unroll
    for (int i = 0; i < 4; ++i)
#pragma unroll
      for (int j = 0; j < 4; ++j) acc[i][j] = zero;
  }

  auto compute = [&](int buf) {
    // A fragment: 16x32 f16; lane<16 -> K 0-7 & 16-23, lane>=16 -> +8
    const int ar  = wm * 64 + (lane & 15);
    const int akb = (lane >> 4) << 3;          // 0 / 8
    v16h af[4];
#pragma unroll
    for (int i = 0; i < 4; ++i) {
      const _Float16* p = &sA[buf][(ar + i * 16) * LDT + akb];
      v8h lo = *(const v8h*)p;            // K kb..kb+7
      v8h hi = *(const v8h*)(p + 16);     // K kb+16..kb+23
      af[i] = __builtin_shufflevector(lo, hi, 0,1,2,3,4,5,6,7,8,9,10,11,12,13,14,15);
    }
    // B fragment: 32x16 f16; lane<16 -> K 0-7 & 16-23, lane>=16 -> +8
    const int br  = wn * 64 + (lane & 15);
    const int bkb = (lane >> 4) << 3;          // 0 / 8
    v16h bf[4];
#pragma unroll
    for (int j = 0; j < 4; ++j) {
      const _Float16* p = &sB[buf][(br + j * 16) * LDT + bkb];
      v8h lo = *(const v8h*)p;            // K kb..kb+7
      v8h hi = *(const v8h*)(p + 16);     // K kb+16..kb+23
      bf[j] = __builtin_shufflevector(lo, hi, 0,1,2,3,4,5,6,7,8,9,10,11,12,13,14,15);
    }
#pragma unroll
    for (int i = 0; i < 4; ++i)
#pragma unroll
      for (int j = 0; j < 4; ++j)
        acc[i][j] = __builtin_amdgcn_wmma_f32_16x16x32_f16(
            false, af[i], false, bf[j], (short)0, acc[i][j], false, false);
  };

  // software pipeline: prologue fill, then overlap global loads with WMMA
  loadG(0);
  storeL(0);
  __syncthreads();
  for (int s = 0; s < S; ++s) {
    const int buf = s & 1;
    if (s + 1 < S) loadG(s + 1);     // issue next stage's vmem loads
    compute(buf);                    // 16 WMMAs from LDS
    if (s + 1 < S) storeL(buf ^ 1);  // convert + fill other buffer
    __syncthreads();
  }

  // epilogue: C layout = VGPR r -> row r (lanes 0-15) / row r+8 (lanes 16-31)
#pragma unroll
  for (int j = 0; j < 4; ++j) {
    const int n  = bn + wn * 64 + j * 16 + (lane & 15);
    const float bv = bias[n];
#pragma unroll
    for (int i = 0; i < 4; ++i) {
      const int m0 = bm + wm * 64 + i * 16 + ((lane >> 4) << 3);
#pragma unroll
      for (int r = 0; r < 8; ++r)
        out[(size_t)(m0 + r) * N + n] = acc[i][j][r] + bv;
    }
  }
}

extern "C" void kernel_launch(void* const* d_in, const int* in_sizes, int n_in,
                              void* d_out, int out_size, void* d_ws, size_t ws_size,
                              hipStream_t stream) {
  const float* x    = (const float*)d_in[0];
  const int*   qw   = (const int*)d_in[1];
  const float* sc   = (const float*)d_in[2];
  const int*   qz   = (const int*)d_in[3];
  const float* bias = (const float*)d_in[4];
  float* out = (float*)d_out;

  const int N = in_sizes[4];                                  // bias: [N]
  const int K = (int)(((long long)in_sizes[2] * GSZ) / N);    // scales: [K/G, N]
  const int M = in_sizes[0] / K;                              // x: [M, K]

  dim3 grid(N / BN, M / BM);   // 11008/128=86, 4096/256=16 -> exact tiling
  awq_gemm_f16_wmma<<<grid, NTHREADS, 0, stream>>>(x, qw, sc, qz, bias, out, M, K, N);
}
